// GramLoss_24077586661772
// MI455X (gfx1250) — matrix operand order repaired and checked by reading, not compiled, not run on hardware
//
#include <hip/hip_runtime.h>
#include <hip/hip_bf16.h>

typedef __attribute__((ext_vector_type(16))) _Float16 v16h;
typedef __attribute__((ext_vector_type(8)))  _Float16 v8h;
typedef __attribute__((ext_vector_type(8)))  float    v8f;

#define B_      32
#define N_      1024
#define D_      768
#define TM      64
#define TN      64
#define KC      32
#define NK      (D_ / KC)       // 24 K-steps
#define LDT     40              // padded LDS row stride in halves (80B, 16B-aligned)
#define TS_     (TM * LDT)      // halves per tile (2560)
#define NTILE   (N_ / TM)       // 16
#define NPAIRS  (NTILE * (NTILE + 1) / 2)   // 136 upper-triangular tile pairs
#define NBLOCKS (B_ * NPAIRS)               // 4352

#if defined(__AMDGCN__) && __has_builtin(__builtin_amdgcn_global_load_async_to_lds_b128)
#define USE_ASYNC_LDS 1
#define NBUF 2
#else
#define USE_ASYNC_LDS 0
#define NBUF 1
#endif

#if USE_ASYNC_LDS
// builtin signature (probe-confirmed): arg0 = global int4-vector*, arg1 = LDS int4-vector*
typedef int v4i __attribute__((vector_size(16)));
typedef __attribute__((address_space(1))) v4i as1_v4i;
typedef __attribute__((address_space(3))) v4i as3_v4i;
#endif

__device__ __forceinline__ void wait_async0() {
#if USE_ASYNC_LDS
#if __has_builtin(__builtin_amdgcn_s_wait_asynccnt)
    __builtin_amdgcn_s_wait_asynccnt(0);
#else
    asm volatile("s_wait_asynccnt 0x0" ::: "memory");
#endif
#endif
}

// ---------------------------------------------------------------------------
// Kernel 1: L2-normalize rows (fp32 in) -> f16 out (torch F.normalize semantics)
// ---------------------------------------------------------------------------
__global__ __launch_bounds__(256) void gram_normalize_kernel(
    const float* __restrict__ s_in, const float* __restrict__ t_in,
    _Float16* __restrict__ s_out, _Float16* __restrict__ t_out)
{
    const int row = blockIdx.x;
    const float*   in  = (blockIdx.y == 0) ? s_in  : t_in;
    _Float16*      out = (blockIdx.y == 0) ? s_out : t_out;
    const int t = threadIdx.x;

    const float* p = in + (size_t)row * D_ + 3 * t;
    float x0 = p[0], x1 = p[1], x2 = p[2];

    __shared__ float red[256];
    red[t] = x0 * x0 + x1 * x1 + x2 * x2;
    __syncthreads();
    #pragma unroll
    for (int off = 128; off > 0; off >>= 1) {
        if (t < off) red[t] += red[t + off];
        __syncthreads();
    }
    __shared__ float inv_s;
    if (t == 0) inv_s = 1.0f / fmaxf(sqrtf(red[0]), 1e-12f);
    __syncthreads();
    const float inv = inv_s;

    _Float16* q = out + (size_t)row * D_ + 3 * t;
    q[0] = (_Float16)(x0 * inv);
    q[1] = (_Float16)(x1 * inv);
    q[2] = (_Float16)(x2 * inv);
}

// ---------------------------------------------------------------------------
// Fragment gather: combine two 16B LDS chunks into one v16h WMMA operand
// ---------------------------------------------------------------------------
__device__ __forceinline__ v16h frag2(const _Float16* p0, const _Float16* p1)
{
    v8h lo = *(const v8h*)p0;
    v8h hi = *(const v8h*)p1;
    v16h f;
    #pragma unroll
    for (int i = 0; i < 8; ++i) { f[i] = lo[i]; f[i + 8] = hi[i]; }
    return f;
}

#define WMMA_F16(A, Bm, C) \
    __builtin_amdgcn_wmma_f32_16x16x32_f16(false, A, false, Bm, (short)0, C, false, false)

// One K-step for a 32x32 wave tile (both matrices): 8 frag gathers, 8 WMMAs.
// a0/a1: offsets of the two A-frag rows in the s-A tile; b0/b1: offsets in the
// s-B tile (already include +TS_). The t-matrix tiles sit +2*TS_ further.
__device__ __forceinline__ void gram_step8(
    const _Float16* tb,
    int a0, int a1, int b0, int b1,
    v8f& s00, v8f& s01, v8f& s10, v8f& s11,
    v8f& t00, v8f& t01, v8f& t10, v8f& t11)
{
    v16h A0 = frag2(tb + a0, tb + a0 + 8);
    v16h A1 = frag2(tb + a1, tb + a1 + 8);
    v16h B0 = frag2(tb + b0, tb + b0 + 8);
    v16h B1 = frag2(tb + b1, tb + b1 + 8);
    s00 = WMMA_F16(A0, B0, s00);
    s01 = WMMA_F16(A0, B1, s01);
    s10 = WMMA_F16(A1, B0, s10);
    s11 = WMMA_F16(A1, B1, s11);
    A0 = frag2(tb + a0 + 2 * TS_, tb + a0 + 2 * TS_ + 8);
    A1 = frag2(tb + a1 + 2 * TS_, tb + a1 + 2 * TS_ + 8);
    B0 = frag2(tb + b0 + 2 * TS_, tb + b0 + 2 * TS_ + 8);
    B1 = frag2(tb + b1 + 2 * TS_, tb + b1 + 2 * TS_ + 8);
    t00 = WMMA_F16(A0, B0, t00);
    t01 = WMMA_F16(A0, B1, t01);
    t10 = WMMA_F16(A1, B0, t10);
    t11 = WMMA_F16(A1, B1, t11);
}

// ---------------------------------------------------------------------------
// Kernel 2: per-(batch, 64x64 tile-pair) Gram diff. 4 waves, 32x32 per wave.
// ---------------------------------------------------------------------------
__global__ __launch_bounds__(128) void gram_diff_tiles_kernel(
    const _Float16* __restrict__ sH, const _Float16* __restrict__ tH,
    float* __restrict__ partials)
{
    // [buf][tile: As,Bs,At,Bt][TM*LDT]
    __shared__ _Float16 tiles[NBUF * 4 * TS_];
    __shared__ float    red[128];

    const int tid   = threadIdx.x;      // 0..127
    const int bid   = blockIdx.x;
    const int batch = bid / NPAIRS;

    // map residual index -> upper-triangular (mt, nt), mt <= nt (uniform per block)
    int r = bid % NPAIRS;
    int mt = 0;
    while (r >= NTILE - mt) { r -= NTILE - mt; ++mt; }
    const int nt = mt + r;
    const float w = (mt == nt) ? 1.0f : 2.0f;   // off-diagonal tiles counted twice

    const size_t batch_off = (size_t)batch * N_ * D_;
    const int m0 = mt * TM, n0 = nt * TN;

    // staging: 128 threads x 2 chunks per tile; chunk = 16B (8 halves)
    const int lrow = tid >> 2;                 // 0..31 (second chunk: +32)
    const int lcol = (tid & 3) * 8;            // half-offset 0/8/16/24
    const size_t a_off = batch_off + (size_t)(m0 + lrow) * D_ + lcol;
    const size_t b_off = batch_off + (size_t)(n0 + lrow) * D_ + lcol;
    const int lds_off = lrow * LDT + lcol;     // second chunk: +32*LDT

    // wave tiling: 2(M) x 2(N) waves; each wave owns 32x32 (2x2 of 16x16 tiles)
    const int lane = tid & 31;
    const int wave = tid >> 5;                 // 0..3
    const int wm = (wave >> 1) * 32;           // 0 or 32
    const int wn = (wave & 1) * 32;            // 0 or 32

    // 16-bit operand k order used here (contiguous k = 16h + s, same for A and B; wave32):
    //   A: lanes 0-15 -> row M=lane, elems 0..15 = K0..15
    //      lanes 16-31-> row M=lane-16, elems 0..15 = K16..31
    //   B: lanes 0-15 -> col N=lane, elems 0..15 = K0..15 ; lanes 16-31 -> K16..31
    const int fr   = lane & 15;
    const int a_kb = (lane < 16) ? 0 : 16;
    const int b_kb = (lane < 16) ? 0 : 16;
    const int aoff0 = (wm + fr) * LDT + a_kb;              // s-A tile (tile 0)
    const int aoff1 = aoff0 + 16 * LDT;
    const int boff0 = TS_ + (wn + fr) * LDT + b_kb;        // s-B tile (tile 1)
    const int boff1 = boff0 + 16 * LDT;

    v8f s00 = {}, s01 = {}, s10 = {}, s11 = {};
    v8f t00 = {}, t01 = {}, t10 = {}, t11 = {};

#if USE_ASYNC_LDS
    // ---- CDNA5 async global->LDS pipeline: 2 LDS buffers, 1 barrier/K-step ----
    auto issue = [&](int buf, int k0) {
        _Float16* dst = &tiles[(buf * 4) * TS_];
        const _Float16* gsa = sH + a_off + k0;
        const _Float16* gsb = sH + b_off + k0;
        const _Float16* gta = tH + a_off + k0;
        const _Float16* gtb = tH + b_off + k0;
        __builtin_amdgcn_global_load_async_to_lds_b128(
            (as1_v4i*)gsa,             (as3_v4i*)(dst + 0 * TS_ + lds_off), 0, 0);
        __builtin_amdgcn_global_load_async_to_lds_b128(
            (as1_v4i*)(gsa + 32 * D_), (as3_v4i*)(dst + 0 * TS_ + lds_off + 32 * LDT), 0, 0);
        __builtin_amdgcn_global_load_async_to_lds_b128(
            (as1_v4i*)gsb,             (as3_v4i*)(dst + 1 * TS_ + lds_off), 0, 0);
        __builtin_amdgcn_global_load_async_to_lds_b128(
            (as1_v4i*)(gsb + 32 * D_), (as3_v4i*)(dst + 1 * TS_ + lds_off + 32 * LDT), 0, 0);
        __builtin_amdgcn_global_load_async_to_lds_b128(
            (as1_v4i*)gta,             (as3_v4i*)(dst + 2 * TS_ + lds_off), 0, 0);
        __builtin_amdgcn_global_load_async_to_lds_b128(
            (as1_v4i*)(gta + 32 * D_), (as3_v4i*)(dst + 2 * TS_ + lds_off + 32 * LDT), 0, 0);
        __builtin_amdgcn_global_load_async_to_lds_b128(
            (as1_v4i*)gtb,             (as3_v4i*)(dst + 3 * TS_ + lds_off), 0, 0);
        __builtin_amdgcn_global_load_async_to_lds_b128(
            (as1_v4i*)(gtb + 32 * D_), (as3_v4i*)(dst + 3 * TS_ + lds_off + 32 * LDT), 0, 0);
    };

    issue(0, 0);
    for (int i = 0; i < NK; ++i) {
        const int cur = i & 1;
        wait_async0();          // this wave's writes to buf[cur] have landed
        __syncthreads();        // everyone's writes landed; prev reads of buf[cur^1] done
        if (i + 1 < NK) issue(cur ^ 1, (i + 1) * KC);   // overlaps with compute below
        gram_step8(&tiles[(cur * 4) * TS_], aoff0, aoff1, boff0, boff1,
                   s00, s01, s10, s11, t00, t01, t10, t11);
    }
#else
    // ---- fallback: register-prefetch pipeline (loads stay in flight over WMMAs)
    uint4 r0 = *(const uint4*)(sH + a_off);
    uint4 r1 = *(const uint4*)(sH + a_off + 32 * D_);
    uint4 r2 = *(const uint4*)(sH + b_off);
    uint4 r3 = *(const uint4*)(sH + b_off + 32 * D_);
    uint4 r4 = *(const uint4*)(tH + a_off);
    uint4 r5 = *(const uint4*)(tH + a_off + 32 * D_);
    uint4 r6 = *(const uint4*)(tH + b_off);
    uint4 r7 = *(const uint4*)(tH + b_off + 32 * D_);
    for (int i = 0; i < NK; ++i) {
        __syncthreads();        // previous iteration's LDS reads done
        *(uint4*)&tiles[0 * TS_ + lds_off] = r0;
        *(uint4*)&tiles[0 * TS_ + lds_off + 32 * LDT] = r1;
        *(uint4*)&tiles[1 * TS_ + lds_off] = r2;
        *(uint4*)&tiles[1 * TS_ + lds_off + 32 * LDT] = r3;
        *(uint4*)&tiles[2 * TS_ + lds_off] = r4;
        *(uint4*)&tiles[2 * TS_ + lds_off + 32 * LDT] = r5;
        *(uint4*)&tiles[3 * TS_ + lds_off] = r6;
        *(uint4*)&tiles[3 * TS_ + lds_off + 32 * LDT] = r7;
        __syncthreads();        // tiles visible to all waves
        if (i + 1 < NK) {
            const int k0 = (i + 1) * KC;
            r0 = *(const uint4*)(sH + a_off + k0);
            r1 = *(const uint4*)(sH + a_off + 32 * D_ + k0);
            r2 = *(const uint4*)(sH + b_off + k0);
            r3 = *(const uint4*)(sH + b_off + 32 * D_ + k0);
            r4 = *(const uint4*)(tH + a_off + k0);
            r5 = *(const uint4*)(tH + a_off + 32 * D_ + k0);
            r6 = *(const uint4*)(tH + b_off + k0);
            r7 = *(const uint4*)(tH + b_off + 32 * D_ + k0);
        }
        gram_step8(tiles, aoff0, aoff1, boff0, boff1,
                   s00, s01, s10, s11, t00, t01, t10, t11);
    }
#endif

    // epilogue: relu, diff, square, weighted per-lane partial (32 elems/lane)
    float lsum = 0.0f;
    #pragma unroll
    for (int i = 0; i < 8; ++i) {
        float d0 = fmaxf(s00[i], 0.0f) - fmaxf(t00[i], 0.0f);
        float d1 = fmaxf(s01[i], 0.0f) - fmaxf(t01[i], 0.0f);
        float d2 = fmaxf(s10[i], 0.0f) - fmaxf(t10[i], 0.0f);
        float d3 = fmaxf(s11[i], 0.0f) - fmaxf(t11[i], 0.0f);
        lsum += d0 * d0 + d1 * d1 + d2 * d2 + d3 * d3;
    }
    lsum *= w;

    __syncthreads();
    red[tid] = lsum;
    __syncthreads();
    #pragma unroll
    for (int off = 64; off > 0; off >>= 1) {
        if (tid < off) red[tid] += red[tid + off];
        __syncthreads();
    }
    if (tid == 0) partials[bid] = red[0];
}

// ---------------------------------------------------------------------------
// Kernel 3: deterministic final reduction + mean scale
// ---------------------------------------------------------------------------
__global__ __launch_bounds__(256) void gram_final_reduce_kernel(
    const float* __restrict__ partials, float* __restrict__ out)
{
    float s = 0.0f;
    for (int i = threadIdx.x; i < NBLOCKS; i += 256) s += partials[i];
    __shared__ float red[256];
    red[threadIdx.x] = s;
    __syncthreads();
    #pragma unroll
    for (int off = 128; off > 0; off >>= 1) {
        if (threadIdx.x < off) red[threadIdx.x] += red[threadIdx.x + off];
        __syncthreads();
    }
    if (threadIdx.x == 0)
        out[0] = red[0] * (1.0f / ((float)B_ * (float)N_ * (float)N_));
}

extern "C" void kernel_launch(void* const* d_in, const int* in_sizes, int n_in,
                              void* d_out, int out_size, void* d_ws, size_t ws_size,
                              hipStream_t stream) {
    const float* s_in = (const float*)d_in[0];
    const float* t_in = (const float*)d_in[1];

    const size_t elems = (size_t)B_ * N_ * D_;
    _Float16* sH = (_Float16*)d_ws;            // 48 MB
    _Float16* tH = sH + elems;                 // 48 MB
    float* partials = (float*)(tH + elems);    // 4352 floats

    gram_normalize_kernel<<<dim3(B_ * N_, 2), 256, 0, stream>>>(s_in, t_in, sH, tH);
    gram_diff_tiles_kernel<<<NBLOCKS, 128, 0, stream>>>(sH, tH, partials);
    gram_final_reduce_kernel<<<1, 256, 0, stream>>>(partials, (float*)d_out);
}
